// DCN_87024627351781
// MI455X (gfx1250) — matrix operand order repaired and checked by reading, not compiled, not run on hardware
//
#include <hip/hip_runtime.h>

typedef __attribute__((ext_vector_type(16))) _Float16 v16h;
typedef __attribute__((ext_vector_type(8)))  float    v8f;

#define BATCH 4
#define Hh    96
#define Ww    96
#define Cc    256
#define Ff    256
#define KKt   9
#define Mtot  (BATCH * Hh * Ww)   // 36864
#define MT    64                  // M-tile rows per block
#define KCH   32                  // channels per WMMA K step
#define NSTEP (KKt * (Cc / KCH))  // 72 K-steps

// ---------------- kernel 1: offset/mask 3x3 conv -> ws.om [M][27] ----------
// channels 0..8 = dy, 9..17 = dx, 18..26 = sigmoid(mask)
__global__ __launch_bounds__(256) void dcn_offmask(
    const float* __restrict__ x, const float* __restrict__ w_off,
    const float* __restrict__ b_off, float* __restrict__ om) {
  int idx = blockIdx.x * 256 + threadIdx.x;
  if (idx >= Mtot * 27) return;
  int oc  = idx % 27;
  int pix = idx / 27;
  int wx = pix % Ww;
  int hy = (pix / Ww) % Hh;
  int b  = pix / (Ww * Hh);
  float acc = b_off[oc];
  for (int ky = 0; ky < 3; ++ky) {
    int yy = hy + ky - 1;
    if (yy < 0 || yy >= Hh) continue;
    for (int kx = 0; kx < 3; ++kx) {
      int xx = wx + kx - 1;
      if (xx < 0 || xx >= Ww) continue;
      const float* xp = x + (size_t)((b * Hh + yy) * Ww + xx) * Cc;
      const float* wp = w_off + (size_t)((ky * 3 + kx) * Cc) * 27 + oc;
      #pragma unroll 4
      for (int c = 0; c < Cc; ++c) acc = fmaf(xp[c], wp[c * 27], acc);
    }
  }
  if (oc >= 18) acc = 1.0f / (1.0f + __expf(-acc));
  om[(size_t)pix * 27 + oc] = acc;
}

// ---------------- kernel 2: w_conv f32 -> f16 (B operand) ------------------
__global__ __launch_bounds__(256) void dcn_cvtw(const float* __restrict__ w,
                                                _Float16* __restrict__ wh) {
  int i = blockIdx.x * 256 + threadIdx.x;  // exactly KK*C*F threads
  wh[i] = (_Float16)w[i];
}

// ---------------- kernel 3: fused deform-sample + WMMA GEMM ----------------
// out[M, F] = sum_k A[M, k] * Bw[k, F];  A is built on the fly into a
// double-buffered LDS tile (f16), consume-before-fill software pipeline.
__device__ __forceinline__ void fill_tile(
    const float* __restrict__ x,
    const int   (*s_off)[MT][4], const float (*s_wgt)[MT][4],
    _Float16 (*buf)[KCH + 8], int step, int tid) {
  int kk = step >> 3;                 // tap
  int cb = ((step & 7) << 5) * 4;     // channel chunk base, in BYTES
  #pragma unroll
  for (int e = tid; e < MT * (KCH / 4); e += 256) {  // 512 float4 elems
    int p  = e >> 3;                  // pixel row in tile
    int c4 = (e & 7) * 4;             // channel quad
    const int*   of = s_off[kk][p];   // corner byte offsets
    const float* wg = s_wgt[kk][p];
    float ax = 0.0f, ay = 0.0f, az = 0.0f, aw = 0.0f;
    #pragma unroll
    for (int r = 0; r < 4; ++r) {
      // 32-bit unsigned offset + uniform base -> saddr global_load_b128
      unsigned ob = (unsigned)of[r] + (unsigned)(cb + c4 * 4);
      float4 xv = *(const float4*)((const char*)x + ob);
      ax = fmaf(wg[r], xv.x, ax);
      ay = fmaf(wg[r], xv.y, ay);
      az = fmaf(wg[r], xv.z, az);
      aw = fmaf(wg[r], xv.w, aw);
    }
    union { _Float16 h[4]; unsigned long long q; } pk;
    pk.h[0] = (_Float16)ax;
    pk.h[1] = (_Float16)ay;
    pk.h[2] = (_Float16)az;
    pk.h[3] = (_Float16)aw;
    *(unsigned long long*)(&buf[p][c4]) = pk.q;  // ds_store_b64
  }
}

__global__ __launch_bounds__(256) void dcn_main(
    const float* __restrict__ x, const float* __restrict__ om,
    const _Float16* __restrict__ wh, const float* __restrict__ b_conv,
    float* __restrict__ out) {
  __shared__ int      s_off[KKt][MT][4];      // corner base offsets (BYTES)
  __shared__ float    s_wgt[KKt][MT][4];      // bilinear weight * mask (0 if OOB)
  __shared__ _Float16 s_a[2][MT][KCH + 8];    // double-buffered A tile

  const int tid  = threadIdx.x;
  const int lane = tid & 31;
  const int wv   = tid >> 5;     // 0..7 (wave32)
  const int mp   = wv & 1;       // m sub-tile pair: rows (mp*2..mp*2+1)*16
  const int ng   = wv >> 1;      // 0..3: n tiles ng*4 .. ng*4+3
  const int m0   = blockIdx.x * MT;

  // ---- per (pixel, tap) sampling parameters ----
  for (int i = tid; i < MT * KKt; i += 256) {
    int p   = i / KKt;
    int kk  = i % KKt;
    int pix = m0 + p;
    int wxp = pix % Ww;
    int hyp = (pix / Ww) % Hh;
    int b   = pix / (Ww * Hh);
    const float* omp = om + (size_t)pix * 27;
    float dy = omp[kk];
    float dx = omp[9 + kk];
    float mk = omp[18 + kk];
    float ys = (float)hyp + (float)(kk / 3 - 1) + dy;
    float xs = (float)wxp + (float)(kk % 3 - 1) + dx;
    float y0f = floorf(ys), x0f = floorf(xs);
    float wy1 = ys - y0f, wx1 = xs - x0f;
    float wy0 = 1.0f - wy1, wx0 = 1.0f - wx1;
    int y0 = (int)y0f, x0 = (int)x0f;
    #pragma unroll
    for (int r = 0; r < 4; ++r) {
      int cy = y0 + (r >> 1);
      int cx = x0 + (r & 1);
      bool valid = (cy >= 0) && (cy < Hh) && (cx >= 0) && (cx < Ww);
      int cyc = min(max(cy, 0), Hh - 1);
      int cxc = min(max(cx, 0), Ww - 1);
      s_off[kk][p][r] = (((b * Hh + cyc) * Ww + cxc) * Cc) * 4;  // bytes
      float wgt = ((r >> 1) ? wy1 : wy0) * ((r & 1) ? wx1 : wx0) * mk;
      s_wgt[kk][p][r] = valid ? wgt : 0.0f;
    }
  }
  __syncthreads();

  v8f zero = {};
  v8f acc[2][4];
  #pragma unroll
  for (int mi = 0; mi < 2; ++mi)
    #pragma unroll
    for (int j = 0; j < 4; ++j) acc[mi][j] = zero;

  // prologue: fill tile for step 0
  fill_tile(x, s_off, s_wgt, s_a[0], 0, tid);

  const int kh = (lane >= 16) ? 4 : 0;   // lanes 16..31: K += 8 halves

  for (int step = 0; step < NSTEP; ++step) {
    __syncthreads();   // A(step) visible; previous consume done

    // ---- consume s_a[step&1]: issue ALL operand loads first ----
    const _Float16* brow = wh + (size_t)(step * 32) * Ff;
    v16h bf[4];
    #pragma unroll
    for (int j = 0; j < 4; ++j)
      #pragma unroll
      for (int s = 0; s < 16; ++s)
        bf[j][s] = brow[(size_t)((s < 8) ? (s + kh * 2) : (16 + (s - 8) + kh * 2)) * Ff + (ng * 4 + j) * 16 + (lane & 15)];

    union { v16h v; unsigned int u[8]; } af[2];
    #pragma unroll
    for (int mi = 0; mi < 2; ++mi) {
      int row = (mp * 2 + mi) * 16 + (lane & 15);
      const unsigned int* ar = (const unsigned int*)(&s_a[step & 1][row][0]);
      #pragma unroll
      for (int i = 0; i < 8; ++i) {
        int k2 = (i < 4) ? (i + kh) : (8 + (i - 4) + kh);
        af[mi].u[i] = ar[k2];
      }
    }

    // keep all loads above, all WMMAs below: no sinking into the MMA chain
    __builtin_amdgcn_sched_barrier(0);

    #pragma unroll
    for (int j = 0; j < 4; ++j) {
      acc[0][j] = __builtin_amdgcn_wmma_f32_16x16x32_f16(
          false, af[0].v, false, bf[j], (short)0, acc[0][j], false, false);
      acc[1][j] = __builtin_amdgcn_wmma_f32_16x16x32_f16(
          false, af[1].v, false, bf[j], (short)0, acc[1][j], false, false);
    }

    // ---- fill next tile (other buffer) while WMMAs drain ----
    if (step + 1 < NSTEP) {
      __builtin_prefetch(
          (const char*)(wh + (size_t)((step + 1) * 32 + lane) * Ff), 0, 1);
      fill_tile(x, s_off, s_wgt, s_a[(step + 1) & 1], step + 1, tid);
    }
  }

  // ---- epilogue: C/D layout (lane 0-15: M=v, lane 16-31: M=v+8) ----
  int col16 = lane & 15;
  #pragma unroll
  for (int mi = 0; mi < 2; ++mi) {
    int rbase = m0 + (mp * 2 + mi) * 16 + ((lane >= 16) ? 8 : 0);
    #pragma unroll
    for (int j = 0; j < 4; ++j) {
      int col = (ng * 4 + j) * 16 + col16;
      float bias = b_conv[col];
      #pragma unroll
      for (int v = 0; v < 8; ++v) {
        out[(size_t)(rbase + v) * Ff + col] = acc[mi][j][v] + bias;
      }
    }
  }
}

extern "C" void kernel_launch(void* const* d_in, const int* in_sizes, int n_in,
                              void* d_out, int out_size, void* d_ws, size_t ws_size,
                              hipStream_t stream) {
  const float* x      = (const float*)d_in[0];
  const float* w_off  = (const float*)d_in[1];
  const float* b_off  = (const float*)d_in[2];
  const float* w_conv = (const float*)d_in[3];
  const float* b_conv = (const float*)d_in[4];
  float* out = (float*)d_out;

  // ws layout: [0, Mtot*27*4) om  | then f16 weights (offset is 32B aligned)
  float*    om = (float*)d_ws;
  _Float16* wh = (_Float16*)((char*)d_ws + (size_t)Mtot * 27 * 4);

  dcn_offmask<<<(Mtot * 27 + 255) / 256, 256, 0, stream>>>(x, w_off, b_off, om);
  dcn_cvtw<<<(KKt * Cc * Ff) / 256, 256, 0, stream>>>(w_conv, wh);
  dcn_main<<<Mtot / MT, 256, 0, stream>>>(x, om, wh, b_conv, out);
}
